// Seq2SeqDecoder_90752658964841
// MI455X (gfx1250) — matrix operand order repaired and checked by reading, not compiled, not run on hardware
//
#include <hip/hip_runtime.h>
#include <hip/hip_bf16.h>

// ---------------- constants (match reference) ----------------
#define BB   32
#define TT   64
#define SS   128
#define DD   768
#define E2D  1024
#define HH   512
#define II   1792          // E2 + D
#define H3   1536          // 3*H
#define CI   2304          // H + E2 + D (fc concat width)

typedef __attribute__((ext_vector_type(16))) _Float16 v16h;
typedef __attribute__((ext_vector_type(8)))  _Float16 v8h;
typedef __attribute__((ext_vector_type(8)))  float    v8f;

// =============================================================
// Generic WMMA GEMM:  C(M x N, f32, row stride ldc) =
//     A(M x K, f16 row-major) * B(K x N, f16 row-major) + bias
// One wave per 16x16 C tile; K-loop of 16x16x32 WMMAs.
// M, N multiples of 16; K multiple of 32.
// =============================================================
__global__ void k_gemm_f16(const _Float16* __restrict__ A,
                           const _Float16* __restrict__ B,
                           const float* __restrict__ bias,
                           float* __restrict__ C,
                           int M, int N, int K, int ldc)
{
    const int wave = (int)((blockIdx.x * blockDim.x + threadIdx.x) >> 5);
    const int lane = (int)(threadIdx.x & 31);
    const int mT = M >> 4, nT = N >> 4;
    if (wave >= mT * nT) return;              // uniform per wave -> EXEC stays all-ones
    const int tm = wave % mT;
    const int tn = wave / mT;

    const int m  = tm * 16 + (lane & 15);     // A row for this lane
    const int g  = lane >> 4;                 // K-half select for A
    const int n0 = tn * 16;

    v8f c = {};
    const _Float16* Arow = A + (size_t)m * K;
    for (int k0 = 0; k0 < K; k0 += 32) {
        const v8h a0 = *(const v8h*)(Arow + k0 + 8 * g);
        const v8h a1 = *(const v8h*)(Arow + k0 + 16 + 8 * g);
        const _Float16* Bcol = B + (size_t)k0 * N + n0 + (lane & 15);
        v16h av, bv;
#pragma unroll
        for (int i = 0; i < 8; ++i) {
            av[i] = a0[i]; av[i + 8] = a1[i];
            bv[i] = Bcol[(size_t)(8 * g + i) * N]; bv[i + 8] = Bcol[(size_t)(16 + 8 * g + i) * N];
        }
        c = __builtin_amdgcn_wmma_f32_16x16x32_f16(
                false, av, false, bv, (short)0, c, false, false);
    }

    const int nc = n0 + (lane & 15);
    const float bb = bias ? bias[nc] : 0.0f;
    const int mBase = tm * 16 + 8 * g;
#pragma unroll
    for (int r = 0; r < 8; ++r)
        C[(size_t)(mBase + r) * ldc + nc] = c[r] + bb;
}

// =============================================================
// Prep kernels
// =============================================================
// Wt[k*N + n] = (f16) W[n*K + k]   (weight transpose + convert)
__global__ void k_trans_cvt(const float* __restrict__ W, _Float16* __restrict__ Wt,
                            int N, int K, int total)
{
    int idx = blockIdx.x * blockDim.x + threadIdx.x;
    if (idx >= total) return;
    int n = idx % N, k = idx / N;
    Wt[idx] = (_Float16)W[(size_t)n * K + k];
}

__global__ void k_cvt(const float* __restrict__ src, _Float16* __restrict__ dst, int total)
{
    int idx = blockIdx.x * blockDim.x + threadIdx.x;
    if (idx < total) dst[idx] = (_Float16)src[idx];
}

// embT[b,t,d] = embed_W[ids[b,t]*mask[b,t], d] * sqrt(D)   (f16)
__global__ void k_embed(const int* __restrict__ ids, const long long* __restrict__ mask,
                        const float* __restrict__ embW, _Float16* __restrict__ embT,
                        float scale, int total)
{
    int idx = blockIdx.x * blockDim.x + threadIdx.x;
    if (idx >= total) return;
    int bt = idx / DD, d = idx % DD;
    long long row = (long long)ids[bt] * mask[bt];
    embT[idx] = (_Float16)(embW[(size_t)row * DD + d] * scale);
}

__global__ void k_init_h(const float* __restrict__ enc_hidden,
                         float* __restrict__ h, _Float16* __restrict__ h16)
{
    int idx = blockIdx.x * blockDim.x + threadIdx.x;   // B*H
    if (idx >= BB * HH) return;
    float v = enc_hidden[idx];
    h[idx] = v;
    h16[idx] = (_Float16)v;
}

// =============================================================
// Attention (per batch element): hWa = h @ Wa_h, scores = tanh(ep+hWa)·val_w,
// softmax over S, weighted = attn @ enc.  Also assembles x=[emb,weighted]
// and the fc-concat slices [.,weighted,emb].  One block per b, 256 threads.
// =============================================================
__global__ void k_attn(const float* __restrict__ ep,      // B*S*H (bias included)
                       const float* __restrict__ h,       // B*H f32
                       const float* __restrict__ attnW,   // (H+E2) x H, rows 0..H-1 = Wa_h
                       const float* __restrict__ valw,    // H
                       const float* __restrict__ enc,     // B*S*E2 f32
                       const _Float16* __restrict__ embT, // B*T*D f16
                       _Float16* __restrict__ xbuf,       // B*II
                       _Float16* __restrict__ cbuf,       // B*CI
                       int t)
{
    __shared__ float sh[HH];
    __shared__ float shWa[HH];
    __shared__ float sval[HH];
    __shared__ float spart[2][SS];
    __shared__ float sattn[SS];
    __shared__ float sred[2];

    const int b = blockIdx.x;
    const int tid = threadIdx.x;   // 0..255

    for (int i = tid; i < HH; i += 256) { sh[i] = h[b * HH + i]; sval[i] = valw[i]; }
    __syncthreads();

    // hWa[n] = sum_k h[k] * Wa_h[k,n]
    for (int n = tid; n < HH; n += 256) {
        float acc = 0.f;
        for (int k = 0; k < HH; ++k) acc += sh[k] * attnW[(size_t)k * HH + n];
        shWa[n] = acc;
    }
    __syncthreads();

    // scores (each s handled by 2 threads, 256-wide halves)
    {
        const int s = tid & (SS - 1);
        const int half = tid >> 7;
        const float* eprow = ep + ((size_t)b * SS + s) * HH + half * 256;
        float acc = 0.f;
        for (int k = 0; k < 256; ++k)
            acc += tanhf(eprow[k] + shWa[half * 256 + k]) * sval[half * 256 + k];
        spart[half][s] = acc;
    }
    __syncthreads();
    if (tid == 0) {
        float mx = -1e30f;
        for (int s = 0; s < SS; ++s) {
            float v = spart[0][s] + spart[1][s];
            sattn[s] = v; if (v > mx) mx = v;
        }
        sred[0] = mx;
    }
    __syncthreads();
    if (tid < SS) sattn[tid] = __expf(sattn[tid] - sred[0]);
    __syncthreads();
    if (tid == 0) {
        float sum = 0.f;
        for (int s = 0; s < SS; ++s) sum += sattn[s];
        sred[1] = 1.f / sum;
    }
    __syncthreads();
    if (tid < SS) sattn[tid] *= sred[1];
    __syncthreads();

    // weighted[e] = sum_s attn[s]*enc[b,s,e]; write f16 into xbuf & cbuf
    {
        const int e = tid * 4;                         // E2 = 256*4
        float4 acc = make_float4(0.f, 0.f, 0.f, 0.f);
        const float* eb = enc + (size_t)b * SS * E2D + e;
        for (int s = 0; s < SS; ++s) {
            float a = sattn[s];
            const float4 v = *(const float4*)(eb + (size_t)s * E2D);
            acc.x += a * v.x; acc.y += a * v.y; acc.z += a * v.z; acc.w += a * v.w;
        }
        _Float16* x = xbuf + (size_t)b * II + DD + e;
        _Float16* c = cbuf + (size_t)b * CI + HH + e;
        x[0] = (_Float16)acc.x; x[1] = (_Float16)acc.y; x[2] = (_Float16)acc.z; x[3] = (_Float16)acc.w;
        c[0] = (_Float16)acc.x; c[1] = (_Float16)acc.y; c[2] = (_Float16)acc.z; c[3] = (_Float16)acc.w;
    }
    // emb slices
    const _Float16* em = embT + ((size_t)b * TT + t) * DD;
    for (int d = tid; d < DD; d += 256) {
        _Float16 v = em[d];
        xbuf[(size_t)b * II + d] = v;
        cbuf[(size_t)b * CI + HH + E2D + d] = v;
    }
}

// =============================================================
// GRU gate combine: r,z,n -> h_new; writes h (f32), h16, cbuf[:, 0:H]
// =============================================================
__global__ void k_gates(const float* __restrict__ gi, const float* __restrict__ gh,
                        float* __restrict__ h, _Float16* __restrict__ h16,
                        _Float16* __restrict__ cbuf)
{
    int idx = blockIdx.x * blockDim.x + threadIdx.x;   // B*H
    if (idx >= BB * HH) return;
    int b = idx >> 9, j = idx & (HH - 1);
    const float* gib = gi + (size_t)b * H3;
    const float* ghb = gh + (size_t)b * H3;
    float r = 1.f / (1.f + __expf(-(gib[j]          + ghb[j])));
    float z = 1.f / (1.f + __expf(-(gib[HH + j]     + ghb[HH + j])));
    float n = tanhf(gib[2 * HH + j] + r * ghb[2 * HH + j]);
    float hn = (1.f - z) * n + z * h[idx];
    h[idx] = hn;
    h16[idx] = (_Float16)hn;
    cbuf[(size_t)b * CI + j] = (_Float16)hn;
}

// =============================================================
// Host-side orchestration
// =============================================================
extern "C" void kernel_launch(void* const* d_in, const int* in_sizes, int n_in,
                              void* d_out, int out_size, void* d_ws, size_t ws_size,
                              hipStream_t stream) {
    const int*       ids        = (const int*)d_in[0];
    const long long* mask       = (const long long*)d_in[1];
    const float*     enc_hidden = (const float*)d_in[2];
    const float*     enc_out    = (const float*)d_in[3];
    const float*     embed_W    = (const float*)d_in[4];
    const float*     attn_W     = (const float*)d_in[5];
    const float*     attn_b     = (const float*)d_in[6];
    const float*     val_w      = (const float*)d_in[7];
    const float*     W_ih       = (const float*)d_in[8];
    const float*     W_hh       = (const float*)d_in[9];
    const float*     b_ih       = (const float*)d_in[10];
    const float*     b_hh       = (const float*)d_in[11];
    const float*     fc_W       = (const float*)d_in[12];
    const float*     fc_b       = (const float*)d_in[13];
    float* out = (float*)d_out;

    // ---- workspace carve ----
    char* p = (char*)d_ws;
    auto alloc = [&](size_t bytes) -> char* {
        char* r = p; p += (bytes + 255) & ~(size_t)255; return r;
    };
    _Float16* Wih_t = (_Float16*)alloc(sizeof(_Float16) * (size_t)II * H3);   // I x 3H
    _Float16* Whh_t = (_Float16*)alloc(sizeof(_Float16) * (size_t)HH * H3);   // H x 3H
    _Float16* fc_t  = (_Float16*)alloc(sizeof(_Float16) * (size_t)CI * DD);   // CI x D
    _Float16* WaE16 = (_Float16*)alloc(sizeof(_Float16) * (size_t)E2D * HH);  // E2 x H
    _Float16* enc16 = (_Float16*)alloc(sizeof(_Float16) * (size_t)BB * SS * E2D);
    float*    ep    = (float*)   alloc(sizeof(float)    * (size_t)BB * SS * HH);
    _Float16* embT  = (_Float16*)alloc(sizeof(_Float16) * (size_t)BB * TT * DD);
    float*    hbuf  = (float*)   alloc(sizeof(float)    * (size_t)BB * HH);
    _Float16* h16   = (_Float16*)alloc(sizeof(_Float16) * (size_t)BB * HH);
    _Float16* xbuf  = (_Float16*)alloc(sizeof(_Float16) * (size_t)BB * II);
    _Float16* cbuf  = (_Float16*)alloc(sizeof(_Float16) * (size_t)BB * CI);
    float*    gi    = (float*)   alloc(sizeof(float)    * (size_t)BB * H3);
    float*    gh    = (float*)   alloc(sizeof(float)    * (size_t)BB * H3);

    const int TPB = 256;
    auto blocks = [](int total) { return (total + 255) / 256; };

    // ---- prep (once per launch; ws is not preserved across calls) ----
    k_trans_cvt<<<blocks(H3 * II), TPB, 0, stream>>>(W_ih, Wih_t, H3, II, H3 * II);
    k_trans_cvt<<<blocks(H3 * HH), TPB, 0, stream>>>(W_hh, Whh_t, H3, HH, H3 * HH);
    k_trans_cvt<<<blocks(DD * CI), TPB, 0, stream>>>(fc_W, fc_t, DD, CI, DD * CI);
    k_cvt<<<blocks(E2D * HH), TPB, 0, stream>>>(attn_W + (size_t)HH * HH, WaE16, E2D * HH);
    k_cvt<<<blocks(BB * SS * E2D), TPB, 0, stream>>>(enc_out, enc16, BB * SS * E2D);
    k_embed<<<blocks(BB * TT * DD), TPB, 0, stream>>>(ids, mask, embed_W, embT,
                                                      27.712812921102035f /*sqrt(768)*/,
                                                      BB * TT * DD);
    k_init_h<<<blocks(BB * HH), TPB, 0, stream>>>(enc_hidden, hbuf, h16);

    // enc_proj = enc16 (4096 x 1024) @ WaE16 (1024 x 512) + attn_b   -> ep
    {
        int tiles = (BB * SS / 16) * (HH / 16);            // 8192
        k_gemm_f16<<<(tiles + 7) / 8, TPB, 0, stream>>>(enc16, WaE16, attn_b, ep,
                                                        BB * SS, HH, E2D, HH);
    }

    // ---- 64 sequential decode steps ----
    for (int t = 0; t < TT; ++t) {
        k_attn<<<BB, TPB, 0, stream>>>(ep, hbuf, attn_W, val_w, enc_out, embT,
                                       xbuf, cbuf, t);
        // gi = x @ W_ih^T + b_ih   (32 x 1536, K=1792)
        {
            int tiles = (BB / 16) * (H3 / 16);             // 192
            k_gemm_f16<<<(tiles + 7) / 8, TPB, 0, stream>>>(xbuf, Wih_t, b_ih, gi,
                                                            BB, H3, II, H3);
        }
        // gh = h @ W_hh^T + b_hh   (32 x 1536, K=512)
        {
            int tiles = (BB / 16) * (H3 / 16);
            k_gemm_f16<<<(tiles + 7) / 8, TPB, 0, stream>>>(h16, Whh_t, b_hh, gh,
                                                            BB, H3, HH, H3);
        }
        k_gates<<<blocks(BB * HH), TPB, 0, stream>>>(gi, gh, hbuf, h16, cbuf);
        // pred = concat @ fc_W^T + fc_b -> out[b, t, :]  (row stride T*D)
        {
            int tiles = (BB / 16) * (DD / 16);             // 96
            k_gemm_f16<<<(tiles + 7) / 8, TPB, 0, stream>>>(cbuf, fc_t, fc_b,
                                                            out + (size_t)t * DD,
                                                            BB, DD, CI, TT * DD);
        }
    }
}
